// WindowAttention_76562087018560
// MI455X (gfx1250) — matrix operand order repaired and checked by reading, not compiled, not run on hardware
//
#include <hip/hip_runtime.h>
#include <hip/hip_fp16.h>

// ---------------- problem constants ----------------
#define SEQ   49            // tokens per window
#define NPAD  64            // padded tokens
#define EMB   512
#define NH    16
#define HD    32
#define NWIN  2048          // total windows
#define NWIMG 64            // windows per image (mask dim 0)
#define M_TOT (NWIN * SEQ)  // 100352 rows (divisible by 256)
#define QKV_F (3 * EMB)     // 1536
#define SCALE_F 0.17677669529663687f  // HD^-0.5

typedef __attribute__((ext_vector_type(16))) _Float16 v16h;
typedef __attribute__((ext_vector_type(8)))  _Float16 v8h;
typedef __attribute__((ext_vector_type(4)))  _Float16 v4h;
typedef __attribute__((ext_vector_type(8)))  float    v8f;
typedef __attribute__((ext_vector_type(4)))  unsigned int v4u;
typedef __attribute__((ext_vector_type(8)))  int      v8i;
typedef __attribute__((ext_vector_type(4)))  int      v4i;

// Tensor Data Mover availability (this toolchain: 6-arg builtin,
// (v4u g0, v8i g1, v4i g2, v4i g3, v8i extra, i32 cpol))
#if defined(__has_builtin)
#if __has_builtin(__builtin_amdgcn_tensor_load_to_lds) && \
    __has_builtin(__builtin_amdgcn_s_wait_tensorcnt)
#define HAVE_TDM 1
#endif
#endif
#ifndef HAVE_TDM
#define HAVE_TDM 0
#endif

// ---- WMMA fragment loaders (CDNA5 16x16x32 f16 layouts, wave32) ----
// A matrix 16x32: lane L<16 holds row M=L, K halves {0..7,16..23};
//                 lane L>=16 holds row M=L-16, K halves {8..15,24..31}.
__device__ __forceinline__ v16h load_frag_a(const _Float16* base, int ld) {
  const int lane = threadIdx.x & 31;
  const _Float16* p = base + (size_t)(lane & 15) * ld + ((lane >> 4) << 3);
  union { v16h v; v8h h[2]; } u;
  u.h[0] = *(const v8h*)(p);
  u.h[1] = *(const v8h*)(p + 16);
  return u.v;
}
// B matrix 32x16: lane holds column N=lane&15; lanes 0-15 K={0..7,16..23},
// lanes 16-31 K={8..15,24..31} (same as A). `base` points at row-major
// B^T, i.e. [N][K] with leading dim ld.
__device__ __forceinline__ v16h load_frag_b(const _Float16* base, int ld) {
  const int lane = threadIdx.x & 31;
  const _Float16* p = base + (size_t)(lane & 15) * ld + ((lane >> 4) << 3);
  union { v16h v; v8h h[2]; } u;
  u.h[0] = *(const v8h*)(p);
  u.h[1] = *(const v8h*)(p + 16);
  return u.v;
}

// ---- TDM 2D tile load: [tile_h rows] x [tile_w elems] of 2-byte data ----
// D# groups built per CDNA5 ISA (count=1, type=2,
// data_size=1 (2B), global_addr = tile start, row stride in elements).
__device__ __forceinline__ void tdm_load_2d_f16(unsigned int lds_byte_addr,
                                                const void* gsrc,
                                                unsigned int tensor_w,
                                                unsigned int tensor_h,
                                                unsigned int row_stride,
                                                unsigned int tile_w,
                                                unsigned int tile_h) {
#if HAVE_TDM
  const unsigned long long ga = (unsigned long long)(uintptr_t)gsrc;
  v4u g0;
  g0[0] = 1u;                                    // count=1, user descriptor
  g0[1] = lds_byte_addr;                         // lds_addr
  g0[2] = (unsigned int)(ga & 0xffffffffu);      // global_addr[31:0]
  g0[3] = (unsigned int)((ga >> 32) & 0x01ffffffu) | (2u << 30);  // type=2
  v8i g1;
  g1[0] = (int)(1u << 16);                       // data_size=1 (2 bytes)
  g1[1] = (int)((tensor_w & 0xffffu) << 16);     // tensor_dim0[15:0]
  g1[2] = (int)(((tensor_w >> 16) & 0xffffu) | ((tensor_h & 0xffffu) << 16));
  g1[3] = (int)(((tensor_h >> 16) & 0xffffu) | ((tile_w & 0xffffu) << 16));
  g1[4] = (int)(tile_h & 0xffffu);               // tile_dim1 (tile_dim2=0)
  g1[5] = (int)row_stride;                       // tensor_dim0_stride[31:0]
  g1[6] = 0;                                     // stride hi / dim1 stride lo
  g1[7] = 0;                                     // dim1 stride hi
  const v4i z4 = {0, 0, 0, 0};
  const v8i z8 = {0, 0, 0, 0, 0, 0, 0, 0};
  __builtin_amdgcn_tensor_load_to_lds(g0, g1, z4, z4, z8, 0);
#else
  (void)lds_byte_addr; (void)gsrc; (void)tensor_w; (void)tensor_h;
  (void)row_stride; (void)tile_w; (void)tile_h;
#endif
}

// ---------------- conversion / table kernels ----------------
__global__ void k_f32_to_f16_x4(const float* __restrict__ src,
                                _Float16* __restrict__ dst, int n4) {
  int i = blockIdx.x * blockDim.x + threadIdx.x;
  if (i >= n4) return;
  float4 f = ((const float4*)src)[i];
  v4h o;
  o.x = (_Float16)f.x; o.y = (_Float16)f.y;
  o.z = (_Float16)f.z; o.w = (_Float16)f.w;
  ((v4h*)dst)[i] = o;
}

// comb[w][h][q][k] = mask[w][q][k] + bias_table[rel_idx[q][k]][h]
__global__ void k_build_comb(const float* __restrict__ mask,
                             const float* __restrict__ bias_table,
                             const int* __restrict__ rel_idx,
                             float* __restrict__ comb) {
  int i = blockIdx.x * blockDim.x + threadIdx.x;
  const int total = NWIMG * NH * SEQ * SEQ;
  if (i >= total) return;
  int k = i % SEQ;
  int t = i / SEQ;
  int q = t % SEQ; t /= SEQ;
  int h = t % NH;
  int w = t / NH;
  comb[i] = mask[((size_t)w * SEQ + q) * SEQ + k] +
            bias_table[(size_t)rel_idx[q * SEQ + k] * NH + h];
}

// ---------------- WMMA GEMM: out[m][f] = sum_e A[m][e]*W[f][e] + bias[f]
// Block: 256 threads = 8 waves, each wave one 32(M) x 64(N) tile.
// B tile (64 rows x 32 halves) double-buffered in LDS via TDM.
__global__ void __launch_bounds__(256)
k_gemm_f16(const _Float16* __restrict__ A, const _Float16* __restrict__ Bw,
           const float* __restrict__ bias, _Float16* __restrict__ outh,
           float* __restrict__ outf, int K, int NOUT) {
  __shared__ __align__(128) _Float16 Bs[2][64 * 32];

  const int wave = threadIdx.x >> 5;
  const int lane = threadIdx.x & 31;
  const int m0 = (blockIdx.x * 8 + wave) * 32;
  const int n0 = blockIdx.y * 64;
  const _Float16* bsrc = Bw + (size_t)n0 * K;

  // stage one B tile into LDS buffer `buf` (k-chunk starting at k0)
  auto stage_b = [&](int buf, int k0) {
#if HAVE_TDM
    if (wave == 0) {
      tdm_load_2d_f16((unsigned int)(uintptr_t)&Bs[buf][0], bsrc + k0,
                      (unsigned int)K, 64u + (unsigned int)n0,
                      (unsigned int)K, 32u, 64u);
    }
#else
    for (int i = threadIdx.x; i < 64 * 4; i += 256) {  // v8h chunks
      const int row = i >> 2, c = (i & 3) * 8;
      *(v8h*)&Bs[buf][row * 32 + c] = *(const v8h*)(bsrc + (size_t)row * K + k0 + c);
    }
#endif
  };
  auto stage_wait = [&]() {
#if HAVE_TDM
    if (wave == 0) __builtin_amdgcn_s_wait_tensorcnt(0);
#endif
    __syncthreads();
  };

  stage_b(0, 0);

  v8f acc[2][4] = {};
  const int nsteps = K / 32;
  for (int s = 0; s < nsteps; ++s) {
    const int buf = s & 1;
    stage_wait();                       // buf ready; prior buf fully consumed
    if (s + 1 < nsteps) stage_b(buf ^ 1, (s + 1) * 32);

    const int k0 = s * 32;
    v16h a0 = load_frag_a(A + (size_t)m0 * K + k0, K);
    v16h a1 = load_frag_a(A + (size_t)(m0 + 16) * K + k0, K);
#pragma unroll
    for (int j = 0; j < 4; ++j) {
      v16h b = load_frag_b(&Bs[buf][(16 * j) * 32], 32);
      acc[0][j] = __builtin_amdgcn_wmma_f32_16x16x32_f16(
          false, a0, false, b, (short)0, acc[0][j], false, false);
      acc[1][j] = __builtin_amdgcn_wmma_f32_16x16x32_f16(
          false, a1, false, b, (short)0, acc[1][j], false, false);
    }
  }

  const int mb = (lane >> 4) << 3;  // 0 or 8
  const int nn = lane & 15;
#pragma unroll
  for (int t = 0; t < 2; ++t) {
#pragma unroll
    for (int j = 0; j < 4; ++j) {
      const int col = n0 + 16 * j + nn;
      const float bv = bias[col];
#pragma unroll
      for (int r = 0; r < 8; ++r) {
        const size_t idx = (size_t)(m0 + 16 * t + mb + r) * NOUT + col;
        const float v = acc[t][j][r] + bv;
        if (outh) outh[idx] = (_Float16)v;
        else      outf[idx] = v;
      }
    }
  }
}

// ---------------- fused window attention: one block per (window, head)
__global__ void __launch_bounds__(128)
k_attn(const _Float16* __restrict__ qkv, const float* __restrict__ comb,
       _Float16* __restrict__ ctx) {
  __shared__ __align__(16) _Float16 Qs[NPAD * HD];   // [token][d]
  __shared__ __align__(16) _Float16 Ks[NPAD * HD];   // [token][d]
  __shared__ __align__(16) _Float16 Vt[HD * NPAD];   // [d][token]
  __shared__ __align__(16) float    Sc[NPAD * NPAD]; // scores f32
  __shared__ __align__(16) _Float16 Pr[NPAD * NPAD]; // probs f16

  const int b = blockIdx.x;
  const int h = blockIdx.y;
  const int tid = threadIdx.x;
  const size_t rowbase = (size_t)b * SEQ;

  // stage Q/K/V for this head into LDS (zero-padded to 64 tokens)
  for (int i = tid; i < NPAD * HD; i += 128) {
    const int m = i >> 5, d = i & 31;
    _Float16 qv = (_Float16)0.f, kv = (_Float16)0.f, vv = (_Float16)0.f;
    if (m < SEQ) {
      const _Float16* row = qkv + (rowbase + m) * QKV_F + h * HD + d;
      qv = row[0];
      kv = row[EMB];
      vv = row[2 * EMB];
    }
    Qs[i] = qv;
    Ks[i] = kv;
    Vt[d * NPAD + m] = vv;
  }
  __syncthreads();

  const int wave = tid >> 5, lane = tid & 31;
  const int m0 = wave * 16;                 // query-row strip per wave
  const int mb = (lane >> 4) << 3, nn = lane & 15;

  // S = Q * K^T  (per wave: 16x64 strip, K-dim = 32 -> 4 WMMAs)
  {
    v16h a = load_frag_a(Qs + m0 * HD, HD);
    v8f acc[4] = {};
#pragma unroll
    for (int j = 0; j < 4; ++j) {
      v16h bf = load_frag_b(Ks + (16 * j) * HD, HD);
      acc[j] = __builtin_amdgcn_wmma_f32_16x16x32_f16(
          false, a, false, bf, (short)0, acc[j], false, false);
    }
    const float* cb =
        comb + (size_t)((b & (NWIMG - 1)) * NH + h) * SEQ * SEQ;
#pragma unroll
    for (int j = 0; j < 4; ++j) {
      const int n = 16 * j + nn;            // key index
#pragma unroll
      for (int r = 0; r < 8; ++r) {
        const int m = m0 + mb + r;          // query index
        float v = -1e30f;
        if (n < SEQ && m < SEQ) v = acc[j][r] * SCALE_F + cb[m * SEQ + n];
        Sc[m * NPAD + n] = v;
      }
    }
  }
  __syncthreads();

  // row softmax (f32), write probs as f16; zero pad rows
  if (tid < NPAD) {
    _Float16* prow = Pr + tid * NPAD;
    if (tid < SEQ) {
      float* srow = Sc + tid * NPAD;
      float mx = -1e30f;
      for (int n = 0; n < SEQ; ++n) mx = fmaxf(mx, srow[n]);
      float sum = 0.f;
      for (int n = 0; n < SEQ; ++n) {
        float ev = __expf(srow[n] - mx);
        srow[n] = ev;
        sum += ev;
      }
      const float inv = 1.f / sum;
      for (int n = 0; n < SEQ; ++n) prow[n] = (_Float16)(srow[n] * inv);
      for (int n = SEQ; n < NPAD; ++n) prow[n] = (_Float16)0.f;
    } else {
      for (int n = 0; n < NPAD; ++n) prow[n] = (_Float16)0.f;
    }
  }
  __syncthreads();

  // O = P * V   (per wave: 16x32 strip, K-dim = 64 -> 2 steps x 2 tiles)
  {
    v16h a0 = load_frag_a(Pr + m0 * NPAD, NPAD);
    v16h a1 = load_frag_a(Pr + m0 * NPAD + 32, NPAD);
    v8f acc[2] = {};
#pragma unroll
    for (int j = 0; j < 2; ++j) {
      v16h b0 = load_frag_b(Vt + (16 * j) * NPAD, NPAD);
      v16h b1 = load_frag_b(Vt + (16 * j) * NPAD + 32, NPAD);
      acc[j] = __builtin_amdgcn_wmma_f32_16x16x32_f16(
          false, a0, false, b0, (short)0, acc[j], false, false);
      acc[j] = __builtin_amdgcn_wmma_f32_16x16x32_f16(
          false, a1, false, b1, (short)0, acc[j], false, false);
    }
#pragma unroll
    for (int j = 0; j < 2; ++j) {
      const int d = 16 * j + nn;
#pragma unroll
      for (int r = 0; r < 8; ++r) {
        const int m = m0 + mb + r;
        if (m < SEQ)
          ctx[(rowbase + m) * EMB + h * HD + d] = (_Float16)acc[j][r];
      }
    }
  }
}

// ---------------- launcher ----------------
extern "C" void kernel_launch(void* const* d_in, const int* in_sizes, int n_in,
                              void* d_out, int out_size, void* d_ws,
                              size_t ws_size, hipStream_t stream) {
  (void)in_sizes; (void)n_in; (void)out_size; (void)ws_size;
  const float* x          = (const float*)d_in[0];
  const float* mask       = (const float*)d_in[1];
  const float* qkv_w      = (const float*)d_in[2];
  const float* qkv_b      = (const float*)d_in[3];
  const float* proj_w     = (const float*)d_in[4];
  const float* proj_b     = (const float*)d_in[5];
  const float* bias_table = (const float*)d_in[6];
  const int*   rel_idx    = (const int*)d_in[7];

  char* ws = (char*)d_ws;
  size_t off = 0;
  _Float16* xh     = (_Float16*)(ws + off); off += (size_t)M_TOT * EMB * 2;
  _Float16* wqkvh  = (_Float16*)(ws + off); off += (size_t)QKV_F * EMB * 2;
  _Float16* wprojh = (_Float16*)(ws + off); off += (size_t)EMB * EMB * 2;
  _Float16* qkvh   = (_Float16*)(ws + off); off += (size_t)M_TOT * QKV_F * 2;
  _Float16* ctxh   = (_Float16*)(ws + off); off += (size_t)M_TOT * EMB * 2;
  float*    comb   = (float*)(ws + off);

  // f32 -> f16 staging
  {
    int n4 = M_TOT * EMB / 4;
    k_f32_to_f16_x4<<<(n4 + 255) / 256, 256, 0, stream>>>(x, xh, n4);
  }
  {
    int n4 = QKV_F * EMB / 4;
    k_f32_to_f16_x4<<<(n4 + 255) / 256, 256, 0, stream>>>(qkv_w, wqkvh, n4);
  }
  {
    int n4 = EMB * EMB / 4;
    k_f32_to_f16_x4<<<(n4 + 255) / 256, 256, 0, stream>>>(proj_w, wprojh, n4);
  }
  // fused (mask + relative position bias) table, L2-resident
  {
    int total = NWIMG * NH * SEQ * SEQ;
    k_build_comb<<<(total + 255) / 256, 256, 0, stream>>>(mask, bias_table,
                                                          rel_idx, comb);
  }

  // QKV projection GEMM: [100352,512] x [1536,512]^T -> f16 [100352,1536]
  k_gemm_f16<<<dim3(M_TOT / 256, QKV_F / 64), 256, 0, stream>>>(
      xh, wqkvh, qkv_b, qkvh, nullptr, EMB, QKV_F);

  // fused attention per (window, head)
  k_attn<<<dim3(NWIN, NH), 128, 0, stream>>>(qkvh, comb, ctxh);

  // output projection GEMM: [100352,512] x [512,512]^T -> f32 d_out
  k_gemm_f16<<<dim3(M_TOT / 256, EMB / 64), 256, 0, stream>>>(
      ctxh, wprojh, proj_b, nullptr, (float*)d_out, EMB, EMB);
}
